// DuplicateUpsampler_88948772700687
// MI455X (gfx1250) — matrix operand order repaired and checked by reading, not compiled, not run on hardware
//
#include <hip/hip_runtime.h>

#define NPTS 16384
#define CDIM 256
#define COUT 256
#define KNN  20
#define RUP  4

typedef __attribute__((ext_vector_type(16))) __bf16 v16bf;
typedef __attribute__((ext_vector_type(8)))  __bf16 v8bf;
typedef __attribute__((ext_vector_type(8)))  float  v8f;

static __device__ __forceinline__ v16bf concat8(v8bf a, v8bf b) {
  return __builtin_shufflevector(a, b, 0,1,2,3,4,5,6,7,8,9,10,11,12,13,14,15);
}

// A-matrix 16x32 bf16 fragment (ISA 7.12.2): lane L holds row M=L&15,
// K = k0..k0+7 and k0+16..k0+23 with k0 = (L<16 ? 0 : 8).
static __device__ __forceinline__ v16bf load_afrag(const __bf16* rowp, int kbase, int lane) {
  int k0 = kbase + ((lane & 16) ? 8 : 0);
  v8bf a0 = *(const v8bf*)(rowp + k0);
  v8bf a1 = *(const v8bf*)(rowp + k0 + 16);
  return concat8(a0, a1);
}

// B-matrix 32x16 bf16 fragment: lane n holds column N=n&15,
// K = k0..k0+7 and k0+16..k0+23 with k0 = (lanes 0-15 ? 0 : 8).
// sB layout: [col][32] bf16 slice of the current K-step.
static __device__ __forceinline__ v16bf load_bfrag(const __bf16* sB, int colbase, int lane) {
  int col = colbase + (lane & 15);
  int k0  = (lane & 16) ? 8 : 0;
  const __bf16* p = sB + col * 32 + k0;
  v8bf b0 = *(const v8bf*)p;
  v8bf b1 = *(const v8bf*)(p + 16);
  return concat8(b0, b1);
}

#define WMMA_BF16(A, B, C) \
  __builtin_amdgcn_wmma_f32_16x16x32_bf16(false, (A), false, (B), (short)0, (C), false, false)

// CDNA5 async global->LDS copy (ASYNCcnt-tracked, bypasses VGPRs).
// NOTE (ISA 08 §4.4): INST_OFFSET is added to BOTH the LDS and global address.
static __device__ __forceinline__ void async_ld16(unsigned lds, unsigned goff,
                                                  const void* base, int ioff) {
  asm volatile("global_load_async_to_lds_b128 %0, %1, %2 offset:%3"
               :: "v"(lds), "v"(goff), "s"(base), "i"(ioff)
               : "memory");
}
static __device__ __forceinline__ void wait_async0() {
  asm volatile("s_wait_asynccnt 0x0" ::: "memory");
}

// ---------------------------------------------------------------------------
// prep: split x into bf16 hi/lo and compute row squared norms
__global__ __launch_bounds__(256)
void prep_x_kernel(const float* __restrict__ x, __bf16* __restrict__ xh,
                   __bf16* __restrict__ xl, float* __restrict__ sq) {
  __shared__ float red[256];
  const int row = blockIdx.x;
  const int t   = threadIdx.x;
  float v = x[(size_t)row * CDIM + t];
  __bf16 h = (__bf16)v;
  float lf = v - (float)h;
  xh[(size_t)row * CDIM + t] = h;
  xl[(size_t)row * CDIM + t] = (__bf16)lf;
  red[t] = v * v;
  __syncthreads();
  for (int s = 128; s > 0; s >>= 1) {
    if (t < s) red[t] += red[t + s];
    __syncthreads();
  }
  if (t == 0) sq[row] = red[0];
}

__global__ __launch_bounds__(256)
void prep_w_kernel(const float* __restrict__ w, __bf16* __restrict__ wh,
                   __bf16* __restrict__ wl) {
  size_t i = (size_t)blockIdx.x * 256 + threadIdx.x;
  float v = w[i];
  __bf16 h = (__bf16)v;
  wh[i] = h;
  wl[i] = (__bf16)(v - (float)h);
}

// Inner product update for one 16x16 sub-tile: 3 WMMAs (hi*hi + hi*lo + lo*hi),
// with a one-deep pipelined B-fragment pair. sched_barrier(0) pins the
// per-subtile schedule so B-fragment liveness stays bounded (no acc spills).
#define SUBTILE_LOOP(sBh, sBl)                                             \
  {                                                                        \
    v16bf Bh = load_bfrag(sBh, 0, lane);                                   \
    v16bf Bl = load_bfrag(sBl, 0, lane);                                   \
    _Pragma("unroll")                                                      \
    for (int st = 0; st < 8; ++st) {                                       \
      v16bf nBh = Bh, nBl = Bl;                                            \
      if (st < 7) {                                                        \
        nBh = load_bfrag(sBh, (st + 1) * 16, lane);                        \
        nBl = load_bfrag(sBl, (st + 1) * 16, lane);                        \
      }                                                                    \
      acc[st] = WMMA_BF16(Ah, Bh, acc[st]);                                \
      acc[st] = WMMA_BF16(Al, Bh, acc[st]);                                \
      acc[st] = WMMA_BF16(Ah, Bl, acc[st]);                                \
      __builtin_amdgcn_sched_barrier(0);                                   \
      Bh = nBh; Bl = nBl;                                                  \
    }                                                                      \
  }

// ---------------------------------------------------------------------------
// Fused distance-GEMM + top-K. Block = 8 waves = 128 rows; loops over 128
// column tiles of 128. Split-bf16 for f32-faithful ranking. B panel
// double-buffered in LDS via async copies; top-20 in registers.
__global__ __launch_bounds__(256, 1)
void knn_kernel(const __bf16* __restrict__ xh, const __bf16* __restrict__ xl,
                const float* __restrict__ sqg, int* __restrict__ idxout) {
  // [0,32768)   : two 16KB B-panel buffers, each {hi 8KB | lo 8KB}
  // [32768,..)  : 512B col norms
  // [33280,..)  : 128x36 f32 dot spill (stride-36 pad)
  __shared__ __align__(16) char smem[32768 + 512 + 128 * 36 * 4];
  float* sSq = (float*)(smem + 32768);
  float* sD  = (float*)(smem + 33280);
  const unsigned smem_lds = (unsigned)(uintptr_t)smem;  // LDS byte address

  const int t    = threadIdx.x;
  const int lane = t & 31;
  const int wave = t >> 5;
  const int rowblk = blockIdx.x * 128;

  const int   myrow = rowblk + (t & 127);
  const float mysq  = sqg[myrow];
  float bd[KNN];
  int   bi[KNN];
#pragma unroll
  for (int k = 0; k < KNN; ++k) { bd[k] = __builtin_inff(); bi[k] = 0; }

  const __bf16* arow_h = xh + (size_t)(rowblk + wave * 16 + (lane & 15)) * CDIM;
  const __bf16* arow_l = xl + (size_t)(rowblk + wave * 16 + (lane & 15)) * CDIM;

  // staging geometry: thread copies 32B of one column's 64B K-slice
  const int      scol = t >> 1;
  const unsigned kh2  = (unsigned)(t & 1) * 32;            // byte offset in K-slice
  const unsigned ldsc = smem_lds + (unsigned)scol * 64 + kh2;

  for (int ct = 0; ct < NPTS / 128; ++ct) {
    __syncthreads();  // protect sSq + buf0 from previous tile's readers
    if (t < 128) sSq[t] = sqg[ct * 128 + t];
    {  // stage ks=0 into buffer 0
      unsigned goff = ((unsigned)(ct * 128 + scol) * CDIM) * 2 + kh2;
      async_ld16(ldsc,        goff, xh, 0);
      async_ld16(ldsc,        goff, xh, 16);
      async_ld16(ldsc + 8192, goff, xl, 0);
      async_ld16(ldsc + 8192, goff, xl, 16);
    }

    v8f acc[8] = {};
    for (int ks = 0; ks < 8; ++ks) {
      wait_async0();     // own async stores to LDS landed
      __syncthreads();   // everyone's landed; previous buffer fully consumed
      const unsigned cur = (unsigned)(ks & 1) * 16384;
      if (ks < 7) {      // prefetch next K-slice into the other buffer
        unsigned nb   = (unsigned)((ks + 1) & 1) * 16384;
        unsigned goff = ((unsigned)(ct * 128 + scol) * CDIM + (ks + 1) * 32) * 2 + kh2;
        async_ld16(ldsc + nb,        goff, xh, 0);
        async_ld16(ldsc + nb,        goff, xh, 16);
        async_ld16(ldsc + nb + 8192, goff, xl, 0);
        async_ld16(ldsc + nb + 8192, goff, xl, 16);
      }
      const __bf16* sBh = (const __bf16*)(smem + cur);
      const __bf16* sBl = (const __bf16*)(smem + cur + 8192);
      v16bf Ah = load_afrag(arow_h, ks * 32, lane);
      v16bf Al = load_afrag(arow_l, ks * 32, lane);
      SUBTILE_LOOP(sBh, sBl)
    }

    // four phases of 32 columns through the small LDS dot buffer
    for (int p = 0; p < 4; ++p) {
      __syncthreads();
      {  // spill 2 sub-tiles (C layout: M = e + (lane>=16 ? 8:0), N = lane&15)
        int mrow = wave * 16 + ((lane & 16) ? 8 : 0);
        int ncol = lane & 15;
#pragma unroll
        for (int st2 = 0; st2 < 2; ++st2) {
          int st = p * 2 + st2;
#pragma unroll
          for (int e = 0; e < 8; ++e)
            sD[(mrow + e) * 36 + st2 * 16 + ncol] = acc[st][e];
        }
      }
      __syncthreads();
      {  // top-K scan: row = t&127, 16 candidate columns per thread
        int r  = t & 127;
        int j0 = (t >> 7) * 16;
        for (int j = j0; j < j0 + 16; ++j) {
          int   cg = ct * 128 + p * 32 + j;
          float d  = mysq + sSq[p * 32 + j] - 2.0f * sD[r * 36 + j];
          if (cg != myrow && d < bd[KNN - 1]) {
            bd[KNN - 1] = d;
            bi[KNN - 1] = cg;
#pragma unroll
            for (int s = KNN - 1; s > 0; --s) {
              if (bd[s] < bd[s - 1]) {
                float td = bd[s]; bd[s] = bd[s - 1]; bd[s - 1] = td;
                int   ti = bi[s]; bi[s] = bi[s - 1]; bi[s - 1] = ti;
              }
            }
          }
        }
      }
    }
  }

  // merge the two per-row half-lists (threads t and t+128) and emit indices
  __syncthreads();
  float* md = (float*)smem;                     // 256*20 f32 = 20 KB
  int*   mi = (int*)(smem + 256 * KNN * 4);     // 256*20 i32 = 20 KB
#pragma unroll
  for (int k = 0; k < KNN; ++k) { md[t * KNN + k] = bd[k]; mi[t * KNN + k] = bi[k]; }
  __syncthreads();
  if (t < 128) {
    const float* d1 = md + t * KNN;
    const float* d2 = md + (t + 128) * KNN;
    const int*   x1 = mi + t * KNN;
    const int*   x2 = mi + (t + 128) * KNN;
    int i1 = 0, i2 = 0;
    int* op = idxout + (size_t)myrow * KNN;
#pragma unroll
    for (int k = 0; k < KNN; ++k) {
      bool take1 = d1[i1] <= d2[i2];
      op[k] = take1 ? x1[i1] : x2[i2];
      if (take1) ++i1; else ++i2;
    }
  }
}

// ---------------------------------------------------------------------------
// y = x @ W^T + b  (split-bf16 WMMA), each row written RUP=4 times.
__global__ __launch_bounds__(256, 1)
void linear_kernel(const __bf16* __restrict__ xh, const __bf16* __restrict__ xl,
                   const __bf16* __restrict__ wh, const __bf16* __restrict__ wl,
                   const float* __restrict__ bias, float* __restrict__ out) {
  __shared__ __align__(16) char smem[32768];  // two 16KB W-panel buffers
  const unsigned smem_lds = (unsigned)(uintptr_t)smem;

  const int t    = threadIdx.x;
  const int lane = t & 31;
  const int wave = t >> 5;
  const int rowblk = blockIdx.x * 128;

  const __bf16* arow_h = xh + (size_t)(rowblk + wave * 16 + (lane & 15)) * CDIM;
  const __bf16* arow_l = xl + (size_t)(rowblk + wave * 16 + (lane & 15)) * CDIM;

  const int      scol = t >> 1;
  const unsigned kh2  = (unsigned)(t & 1) * 32;
  const unsigned ldsc = smem_lds + (unsigned)scol * 64 + kh2;

  for (int ch = 0; ch < 2; ++ch) {  // 2 column halves of 128
    __syncthreads();  // previous half's buffers fully consumed
    {  // stage ks=0 into buffer 0: B[k][n] = W[n][k] since y = x @ W^T
      unsigned goff = ((unsigned)(ch * 128 + scol) * CDIM) * 2 + kh2;
      async_ld16(ldsc,        goff, wh, 0);
      async_ld16(ldsc,        goff, wh, 16);
      async_ld16(ldsc + 8192, goff, wl, 0);
      async_ld16(ldsc + 8192, goff, wl, 16);
    }
    v8f acc[8] = {};
    for (int ks = 0; ks < 8; ++ks) {
      wait_async0();
      __syncthreads();
      const unsigned cur = (unsigned)(ks & 1) * 16384;
      if (ks < 7) {
        unsigned nb   = (unsigned)((ks + 1) & 1) * 16384;
        unsigned goff = ((unsigned)(ch * 128 + scol) * CDIM + (ks + 1) * 32) * 2 + kh2;
        async_ld16(ldsc + nb,        goff, wh, 0);
        async_ld16(ldsc + nb,        goff, wh, 16);
        async_ld16(ldsc + nb + 8192, goff, wl, 0);
        async_ld16(ldsc + nb + 8192, goff, wl, 16);
      }
      const __bf16* sBh = (const __bf16*)(smem + cur);
      const __bf16* sBl = (const __bf16*)(smem + cur + 8192);
      v16bf Ah = load_afrag(arow_h, ks * 32, lane);
      v16bf Al = load_afrag(arow_l, ks * 32, lane);
      SUBTILE_LOOP(sBh, sBl)
    }
    // epilogue: add bias, duplicate each row RUP times ('n c -> (n r) c')
    int mrel = wave * 16 + ((lane & 16) ? 8 : 0);
    int nb   = ch * 128 + (lane & 15);
#pragma unroll
    for (int st = 0; st < 8; ++st) {
      int   n  = nb + st * 16;
      float bv = bias[n];
#pragma unroll
      for (int e = 0; e < 8; ++e) {
        float  v    = acc[st][e] + bv;
        size_t orow = (size_t)(rowblk + mrel + e) * RUP;
#pragma unroll
        for (int r = 0; r < RUP; ++r)
          out[(orow + r) * COUT + n] = v;
      }
    }
  }
}

// ---------------------------------------------------------------------------
extern "C" void kernel_launch(void* const* d_in, const int* in_sizes, int n_in,
                              void* d_out, int out_size, void* d_ws, size_t ws_size,
                              hipStream_t stream) {
  (void)in_sizes; (void)n_in; (void)out_size; (void)ws_size;
  const float* x = (const float*)d_in[0];
  const float* W = (const float*)d_in[1];
  const float* b = (const float*)d_in[2];

  char* ws = (char*)d_ws;
  __bf16* xh = (__bf16*)ws;                                            // 8 MB
  __bf16* xl = (__bf16*)(ws + (size_t)NPTS * CDIM * 2);                // 8 MB
  __bf16* wh = (__bf16*)(ws + (size_t)NPTS * CDIM * 4);                // 128 KB
  __bf16* wl = (__bf16*)(ws + (size_t)NPTS * CDIM * 4 + (size_t)COUT * CDIM * 2);
  float*  sq = (float*) (ws + (size_t)NPTS * CDIM * 4 + (size_t)COUT * CDIM * 4);

  float* out    = (float*)d_out;                                       // [N*R, CO]
  int*   idxout = (int*)((float*)d_out + (size_t)NPTS * RUP * COUT);   // [N, K]

  prep_x_kernel<<<NPTS, 256, 0, stream>>>(x, xh, xl, sq);
  prep_w_kernel<<<(COUT * CDIM) / 256, 256, 0, stream>>>(W, wh, wl);
  linear_kernel<<<NPTS / 128, 256, 0, stream>>>(xh, xl, wh, wl, b, out);
  knn_kernel<<<NPTS / 128, 256, 0, stream>>>(xh, xl, sq, idxout);
}
